// MultiheadAttention_85383949845173
// MI455X (gfx1250) — matrix operand order repaired and checked by reading, not compiled, not run on hardware
//
#include <hip/hip_runtime.h>

#define EMBED 1024
#define SEQ   2048
#define BATCH 2
#define NHEAD 16
#define HDIM  64
#define MROWS (BATCH * SEQ)   // 4096

typedef __attribute__((ext_vector_type(16))) __bf16 v16bf;
typedef __attribute__((ext_vector_type(2)))  __bf16 v2bf;
typedef __attribute__((ext_vector_type(8)))  float  v8f;
typedef __attribute__((ext_vector_type(4)))  int    v4i;

union FragV { uint4 u[2]; v16bf v; };

// native BF16 converts (v_cvt_pk_bf16_f32)
__device__ __forceinline__ unsigned short f2bf(float f) {
  return __builtin_bit_cast(unsigned short, (__bf16)f);
}
__device__ __forceinline__ unsigned pack2(float a, float b) {
  v2bf p; p[0] = (__bf16)a; p[1] = (__bf16)b;
  return __builtin_bit_cast(unsigned, p);
}

__device__ __forceinline__ v8f wmma_bf16(const v16bf a, const v16bf b, v8f c) {
  return __builtin_amdgcn_wmma_f32_16x16x32_bf16(false, a, false, b, (short)0, c,
                                                 false, false);
}

// ---- gfx1250 async global->LDS (ASYNCcnt) with fallback --------------------
#if __has_builtin(__builtin_amdgcn_global_load_async_to_lds_b128)
#define HAVE_ASYNC_LDS 1
typedef __attribute__((address_space(1))) v4i gv4i;
typedef __attribute__((address_space(3))) v4i lv4i;
__device__ __forceinline__ void async_cp16(const void* g, void* l) {
  __builtin_amdgcn_global_load_async_to_lds_b128((gv4i*)g, (lv4i*)l, 0, 0);
}
__device__ __forceinline__ void async_wait() {
  __asm__ volatile("s_wait_asynccnt 0x0" ::: "memory");
}
#else
#define HAVE_ASYNC_LDS 0
__device__ __forceinline__ void async_cp16(const void* g, void* l) {
  *(uint4*)l = *(const uint4*)g;
}
__device__ __forceinline__ void async_wait() {}
#endif

// ---------------------------------------------------------------------------
// Kernel 1: fused QKV projection.  X[4096,1024] @ W[1024,1024] + bias -> bf16
// mode 0 (Q) / 1 (K): out layout [B,H,S,hd]; mode 2 (V): out layout [B,H,hd,S]
// ---------------------------------------------------------------------------
__global__ __launch_bounds__(256)
void proj_qkv_kernel(const float* __restrict__ q_in, const float* __restrict__ k_in,
                     const float* __restrict__ v_in,
                     const float* __restrict__ q_w, const float* __restrict__ k_w,
                     const float* __restrict__ v_w,
                     const float* __restrict__ q_b, const float* __restrict__ k_b,
                     const float* __restrict__ v_b,
                     unsigned short* __restrict__ q_o, unsigned short* __restrict__ k_o,
                     unsigned short* __restrict__ v_o)
{
  const int mode = blockIdx.z;
  const float* X  = (mode == 0) ? q_in : (mode == 1) ? k_in : v_in;
  const float* W  = (mode == 0) ? q_w  : (mode == 1) ? k_w  : v_w;
  const float* Bs = (mode == 0) ? q_b  : (mode == 1) ? k_b  : v_b;
  unsigned short* O = (mode == 0) ? q_o : (mode == 1) ? k_o : v_o;

  __shared__ unsigned short Asm[128 * 32];   // [row][k]  bf16
  __shared__ unsigned short Bsm[128 * 32];   // [n][k]    bf16 (col-major for B frags)

  const int m0   = blockIdx.x * 128;
  const int n0   = blockIdx.y * 128;
  const int tid  = threadIdx.x;
  const int lane = tid & 31;
  const int wave = tid >> 5;
  const int wr   = wave & 3;    // 32-row block
  const int wc   = wave >> 2;   // 64-col block
  const int lh   = lane & 15;
  const int hi   = lane >> 4;

  v8f acc[2][4];
#pragma unroll
  for (int i = 0; i < 2; i++)
#pragma unroll
    for (int n = 0; n < 4; n++)
#pragma unroll
      for (int j = 0; j < 8; j++) acc[i][n][j] = 0.f;

  for (int k0 = 0; k0 < EMBED; k0 += 32) {
    __syncthreads();
    // ---- load phase: issue all 8 global b128 loads into distinct regs ----
    float4 fa[4], fb[4];
#pragma unroll
    for (int i = 0; i < 4; i++) {
      int idx = tid + i * 256;                 // A: 1024 float4 chunks
      int r = idx >> 3, c4 = (idx & 7) << 2;
      fa[i] = *(const float4*)(X + (size_t)(m0 + r) * EMBED + k0 + c4);
    }
#pragma unroll
    for (int i = 0; i < 4; i++) {
      int idx = tid + i * 256;                 // B: 1024 float4 chunks
      int kk = idx >> 5, c4 = (idx & 31) << 2;
      fb[i] = *(const float4*)(W + (size_t)(k0 + kk) * EMBED + n0 + c4);
    }
    // ---- convert/store phase ----
#pragma unroll
    for (int i = 0; i < 4; i++) {
      int idx = tid + i * 256;
      int r = idx >> 3, c4 = (idx & 7) << 2;
      unsigned* dst = (unsigned*)(Asm + r * 32 + c4);
      dst[0] = pack2(fa[i].x, fa[i].y);
      dst[1] = pack2(fa[i].z, fa[i].w);
    }
#pragma unroll
    for (int i = 0; i < 4; i++) {
      int idx = tid + i * 256;
      int kk = idx >> 5, c4 = (idx & 31) << 2;
      Bsm[(c4 + 0) * 32 + kk] = f2bf(fb[i].x);
      Bsm[(c4 + 1) * 32 + kk] = f2bf(fb[i].y);
      Bsm[(c4 + 2) * 32 + kk] = f2bf(fb[i].z);
      Bsm[(c4 + 3) * 32 + kk] = f2bf(fb[i].w);
    }
    // L2 prefetch of next K-step tiles
    if (k0 + 32 < EMBED && tid < 128) {
      __builtin_prefetch(X + (size_t)(m0 + tid) * EMBED + k0 + 32, 0, 1);
      __builtin_prefetch(W + (size_t)(k0 + 32 + (tid >> 2)) * EMBED + n0 + (tid & 3) * 32, 0, 1);
    }
    __syncthreads();

    // A fragments (16x32 bf16, ISA layout: lane<16 -> K0-7 & K16-23)
    FragV a0, a1;
    {
      int r0 = wr * 32 + lh;
      int kb = hi * 8;
      a0.u[0] = *(const uint4*)(Asm + r0 * 32 + kb);
      a0.u[1] = *(const uint4*)(Asm + r0 * 32 + kb + 16);
      int r1 = r0 + 16;
      a1.u[0] = *(const uint4*)(Asm + r1 * 32 + kb);
      a1.u[1] = *(const uint4*)(Asm + r1 * 32 + kb + 16);
    }
    FragV bfr[4];
#pragma unroll
    for (int nt = 0; nt < 4; nt++) {
      int col = wc * 64 + nt * 16 + lh;
      int kb  = hi * 8;                        // B: lanes>=16 hold K8-15 & K24-31
      bfr[nt].u[0] = *(const uint4*)(Bsm + col * 32 + kb);
      bfr[nt].u[1] = *(const uint4*)(Bsm + col * 32 + kb + 16);
    }
#pragma unroll
    for (int nt = 0; nt < 4; nt++) {
      acc[0][nt] = wmma_bf16(a0.v, bfr[nt].v, acc[0][nt]);
      acc[1][nt] = wmma_bf16(a1.v, bfr[nt].v, acc[1][nt]);
    }
  }

  // epilogue: bias + bf16 store in head-layout
#pragma unroll
  for (int i = 0; i < 2; i++)
#pragma unroll
    for (int nt = 0; nt < 4; nt++) {
      int colb = n0 + wc * 64 + nt * 16 + lh;
      float bias = Bs[colb];
      int h = colb >> 6, d = colb & 63;
#pragma unroll
      for (int j = 0; j < 8; j++) {
        int row = m0 + wr * 32 + i * 16 + j + hi * 8;
        int b = row >> 11, s = row & 2047;     // row = b*SEQ + s
        float val = acc[i][nt][j] + bias;
        size_t off;
        if (mode == 2)
          off = ((size_t)(b * NHEAD + h) * HDIM + d) * SEQ + s;      // V^T
        else
          off = ((size_t)(b * NHEAD + h) * SEQ + s) * HDIM + d;      // Q,K
        O[off] = f2bf(val);
      }
    }
}

// ---------------------------------------------------------------------------
// Kernel 2: flash attention.  Q,K:[B,H,S,hd] bf16, V:[B,H,hd,S] bf16
// out: att[B,S,D] bf16.  grid=(S/128, B*H), 8 waves, wave owns 16 q-rows.
// ---------------------------------------------------------------------------
__global__ __launch_bounds__(256)
void attn_kernel(const unsigned short* __restrict__ Q,
                 const unsigned short* __restrict__ K,
                 const unsigned short* __restrict__ V,
                 unsigned short* __restrict__ Oatt)
{
  const int bh  = blockIdx.y;             // b*16 + h
  const int q0  = blockIdx.x * 128;
  const int tid = threadIdx.x;
  const int lane = tid & 31, wave = tid >> 5;
  const int lh = lane & 15, hi = lane >> 4;

  __shared__ unsigned short Ksm[64 * 64];   // [key][d]
  __shared__ unsigned short Vsm[64 * 64];   // [d][key]
  __shared__ unsigned short Psm[128 * 64];  // per-wave P scratch; Q staging first

  // stage Q tile 128x64 into Psm (async), build per-wave A fragments once
#pragma unroll
  for (int i = 0; i < 4; i++) {
    int idx = tid + i * 256;
    int r = idx >> 3, c8 = (idx & 7) << 3;
    async_cp16(Q + ((size_t)bh * SEQ + q0 + r) * HDIM + c8, Psm + r * 64 + c8);
  }
  async_wait();
  __syncthreads();
  FragV aq0, aq1;
  {
    int r = wave * 16 + lh;
    int kb = hi * 8;
    aq0.u[0] = *(const uint4*)(Psm + r * 64 + kb);
    aq0.u[1] = *(const uint4*)(Psm + r * 64 + kb + 16);
    aq1.u[0] = *(const uint4*)(Psm + r * 64 + 32 + kb);
    aq1.u[1] = *(const uint4*)(Psm + r * 64 + 32 + kb + 16);
  }
  __syncthreads();

  v8f o_acc[4];
  float m_i[8], l_i[8];
#pragma unroll
  for (int nt = 0; nt < 4; nt++)
#pragma unroll
    for (int j = 0; j < 8; j++) o_acc[nt][j] = 0.f;
#pragma unroll
  for (int j = 0; j < 8; j++) { m_i[j] = -1e30f; l_i[j] = 0.f; }

  const float sc = 0.125f;   // HEAD_DIM^-0.5

  for (int j0 = 0; j0 < SEQ; j0 += 64) {
    __syncthreads();
    // async load K tile [key][d] and V tile [d][key] (128B rows)
#pragma unroll
    for (int i = 0; i < 2; i++) {
      int idx = tid + i * 256;
      int r = idx >> 3, c8 = (idx & 7) << 3;
      async_cp16(K + ((size_t)bh * SEQ + j0 + r) * HDIM + c8, Ksm + r * 64 + c8);
      async_cp16(V + ((size_t)bh * HDIM + r) * SEQ + j0 + c8, Vsm + r * 64 + c8);
    }
    // L2 prefetch next tiles (global_prefetch_b8)
    if (j0 + 64 < SEQ) {
      __builtin_prefetch(K + ((size_t)bh * SEQ + j0 + 64) * HDIM + tid * 16, 0, 1);
      __builtin_prefetch(V + ((size_t)bh * HDIM + (tid >> 2)) * SEQ + j0 + 64 + (tid & 3) * 16, 0, 1);
    }
    async_wait();
    __syncthreads();

    // scores: S = Q (16x64) @ K^T (64x64) for this wave's 16 rows
    v8f s_[4];
#pragma unroll
    for (int nt = 0; nt < 4; nt++) {
      int col = nt * 16 + lh;
      int kb  = hi * 8;
      FragV b0, b1;
      b0.u[0] = *(const uint4*)(Ksm + col * 64 + kb);
      b0.u[1] = *(const uint4*)(Ksm + col * 64 + kb + 16);
      b1.u[0] = *(const uint4*)(Ksm + col * 64 + 32 + kb);
      b1.u[1] = *(const uint4*)(Ksm + col * 64 + 32 + kb + 16);
      v8f z;
#pragma unroll
      for (int j = 0; j < 8; j++) z[j] = 0.f;
      z = wmma_bf16(aq0.v, b0.v, z);
      z = wmma_bf16(aq1.v, b1.v, z);
      s_[nt] = z;
    }

    // online softmax (rows are j + hi*8; N striped over 16 lanes)
    float alpha[8];
#pragma unroll
    for (int j = 0; j < 8; j++) {
      float t = fmaxf(fmaxf(s_[0][j], s_[1][j]), fmaxf(s_[2][j], s_[3][j])) * sc;
#pragma unroll
      for (int msk = 1; msk < 16; msk <<= 1) t = fmaxf(t, __shfl_xor(t, msk, 32));
      float mn = fmaxf(m_i[j], t);
      alpha[j] = __expf(m_i[j] - mn);
      m_i[j] = mn;
    }
#pragma unroll
    for (int nt = 0; nt < 4; nt++)
#pragma unroll
      for (int j = 0; j < 8; j++)
        s_[nt][j] = __expf(s_[nt][j] * sc - m_i[j]);
#pragma unroll
    for (int j = 0; j < 8; j++) {
      float r = s_[0][j] + s_[1][j] + s_[2][j] + s_[3][j];
#pragma unroll
      for (int msk = 1; msk < 16; msk <<= 1) r += __shfl_xor(r, msk, 32);
      l_i[j] = l_i[j] * alpha[j] + r;
#pragma unroll
      for (int nt = 0; nt < 4; nt++) o_acc[nt][j] *= alpha[j];
    }

    // P (C-layout) -> bf16 -> per-wave LDS -> A-layout fragments
    unsigned short* pw = Psm + wave * 16 * 64;
#pragma unroll
    for (int nt = 0; nt < 4; nt++)
#pragma unroll
      for (int j = 0; j < 8; j++)
        pw[(j + hi * 8) * 64 + nt * 16 + lh] = f2bf(s_[nt][j]);
    __asm__ volatile("" ::: "memory");  // DS ops are in-order per wave
    FragV pa0, pa1;
    {
      int r = lh, kb = hi * 8;
      pa0.u[0] = *(const uint4*)(pw + r * 64 + kb);
      pa0.u[1] = *(const uint4*)(pw + r * 64 + kb + 16);
      pa1.u[0] = *(const uint4*)(pw + r * 64 + 32 + kb);
      pa1.u[1] = *(const uint4*)(pw + r * 64 + 32 + kb + 16);
    }

    // O += P (16x64) @ V (64x64)
#pragma unroll
    for (int nt = 0; nt < 4; nt++) {
      int dc = nt * 16 + lh;
      int kb = hi * 8;
      FragV b0, b1;
      b0.u[0] = *(const uint4*)(Vsm + dc * 64 + kb);
      b0.u[1] = *(const uint4*)(Vsm + dc * 64 + kb + 16);
      b1.u[0] = *(const uint4*)(Vsm + dc * 64 + 32 + kb);
      b1.u[1] = *(const uint4*)(Vsm + dc * 64 + 32 + kb + 16);
      o_acc[nt] = wmma_bf16(pa0.v, b0.v, o_acc[nt]);
      o_acc[nt] = wmma_bf16(pa1.v, b1.v, o_acc[nt]);
    }
  }

  // normalize and store att[b][s][h*64+d] as bf16
  const int h = bh & 15, b = bh >> 4;
#pragma unroll
  for (int nt = 0; nt < 4; nt++) {
    int col = h * 64 + nt * 16 + lh;
#pragma unroll
    for (int j = 0; j < 8; j++) {
      int row = q0 + wave * 16 + j + hi * 8;
      float val = o_acc[nt][j] / l_i[j];
      Oatt[((size_t)b * SEQ + row) * EMBED + col] = f2bf(val);
    }
  }
}

// ---------------------------------------------------------------------------
// Kernel 3: output projection.  att(bf16)[4096,1024] @ o_w + o_b -> fp32 out
// ---------------------------------------------------------------------------
__global__ __launch_bounds__(256)
void out_proj_kernel(const unsigned short* __restrict__ A,
                     const float* __restrict__ W,
                     const float* __restrict__ Bs,
                     float* __restrict__ O)
{
  __shared__ unsigned short Asm[128 * 32];
  __shared__ unsigned short Bsm[128 * 32];

  const int m0 = blockIdx.x * 128;
  const int n0 = blockIdx.y * 128;
  const int tid = threadIdx.x;
  const int lane = tid & 31, wave = tid >> 5;
  const int wr = wave & 3, wc = wave >> 2;
  const int lh = lane & 15, hi = lane >> 4;

  v8f acc[2][4];
#pragma unroll
  for (int i = 0; i < 2; i++)
#pragma unroll
    for (int n = 0; n < 4; n++)
#pragma unroll
      for (int j = 0; j < 8; j++) acc[i][n][j] = 0.f;

  for (int k0 = 0; k0 < EMBED; k0 += 32) {
    __syncthreads();
    // A tile already bf16: async straight copies into LDS
#pragma unroll
    for (int i = 0; i < 2; i++) {
      int idx = tid + i * 256;                // 512 uint4 chunks
      int r = idx >> 2, c8 = (idx & 3) << 3;
      async_cp16(A + (size_t)(m0 + r) * EMBED + k0 + c8, Asm + r * 32 + c8);
    }
    // ---- W load phase then convert/store phase ----
    float4 fb[4];
#pragma unroll
    for (int i = 0; i < 4; i++) {
      int idx = tid + i * 256;
      int kk = idx >> 5, c4 = (idx & 31) << 2;
      fb[i] = *(const float4*)(W + (size_t)(k0 + kk) * EMBED + n0 + c4);
    }
#pragma unroll
    for (int i = 0; i < 4; i++) {
      int idx = tid + i * 256;
      int kk = idx >> 5, c4 = (idx & 31) << 2;
      Bsm[(c4 + 0) * 32 + kk] = f2bf(fb[i].x);
      Bsm[(c4 + 1) * 32 + kk] = f2bf(fb[i].y);
      Bsm[(c4 + 2) * 32 + kk] = f2bf(fb[i].z);
      Bsm[(c4 + 3) * 32 + kk] = f2bf(fb[i].w);
    }
    if (k0 + 32 < EMBED && tid < 128) {
      __builtin_prefetch(W + (size_t)(k0 + 32 + (tid >> 2)) * EMBED + n0 + (tid & 3) * 32, 0, 1);
    }
    async_wait();
    __syncthreads();

    FragV a0, a1;
    {
      int r0 = wr * 32 + lh;
      int kb = hi * 8;
      a0.u[0] = *(const uint4*)(Asm + r0 * 32 + kb);
      a0.u[1] = *(const uint4*)(Asm + r0 * 32 + kb + 16);
      int r1 = r0 + 16;
      a1.u[0] = *(const uint4*)(Asm + r1 * 32 + kb);
      a1.u[1] = *(const uint4*)(Asm + r1 * 32 + kb + 16);
    }
    FragV bfr[4];
#pragma unroll
    for (int nt = 0; nt < 4; nt++) {
      int col = wc * 64 + nt * 16 + lh;
      int kb  = hi * 8;
      bfr[nt].u[0] = *(const uint4*)(Bsm + col * 32 + kb);
      bfr[nt].u[1] = *(const uint4*)(Bsm + col * 32 + kb + 16);
    }
#pragma unroll
    for (int nt = 0; nt < 4; nt++) {
      acc[0][nt] = wmma_bf16(a0.v, bfr[nt].v, acc[0][nt]);
      acc[1][nt] = wmma_bf16(a1.v, bfr[nt].v, acc[1][nt]);
    }
  }

#pragma unroll
  for (int i = 0; i < 2; i++)
#pragma unroll
    for (int nt = 0; nt < 4; nt++) {
      int col = n0 + wc * 64 + nt * 16 + lh;
      float bias = Bs[col];
#pragma unroll
      for (int j = 0; j < 8; j++) {
        int row = m0 + wr * 32 + i * 16 + j + hi * 8;
        O[(size_t)row * EMBED + col] = acc[i][nt][j] + bias;
      }
    }
}

// ---------------------------------------------------------------------------
extern "C" void kernel_launch(void* const* d_in, const int* in_sizes, int n_in,
                              void* d_out, int out_size, void* d_ws, size_t ws_size,
                              hipStream_t stream) {
  const float* query = (const float*)d_in[0];
  const float* key_  = (const float*)d_in[1];
  const float* value = (const float*)d_in[2];
  const float* q_w = (const float*)d_in[3];
  const float* q_b = (const float*)d_in[4];
  const float* k_w = (const float*)d_in[5];
  const float* k_b = (const float*)d_in[6];
  const float* v_w = (const float*)d_in[7];
  const float* v_b = (const float*)d_in[8];
  const float* o_w = (const float*)d_in[9];
  const float* o_b = (const float*)d_in[10];
  float* out = (float*)d_out;

  const size_t NBH = (size_t)BATCH * NHEAD * SEQ * HDIM;  // 4M bf16 elements
  unsigned short* q_buf = (unsigned short*)d_ws;
  unsigned short* k_buf = q_buf + NBH;
  unsigned short* v_buf = k_buf + NBH;
  unsigned short* att   = v_buf + NBH;                    // [B,S,D] bf16

  dim3 blk(256);
  proj_qkv_kernel<<<dim3(MROWS / 128, EMBED / 128, 3), blk, 0, stream>>>(
      query, key_, value, q_w, k_w, v_w, q_b, k_b, v_b, q_buf, k_buf, v_buf);
  attn_kernel<<<dim3(SEQ / 128, BATCH * NHEAD), blk, 0, stream>>>(
      q_buf, k_buf, v_buf, att);
  out_proj_kernel<<<dim3(MROWS / 128, EMBED / 128), blk, 0, stream>>>(
      att, o_w, o_b, out);
}
